// AVT_VQVAE_Encoder_60559038873940
// MI455X (gfx1250) — compile-verified, matrix operand order checked by reading, not run on hardware
//
#include <hip/hip_runtime.h>
#include <hip/hip_bf16.h>
#include <math.h>

// Problem constants (match reference)
#define B_  64
#define T_  256
#define D_  256
#define M_  2048
#define N_  (B_*T_)          // 16384 rows per modality
#define EPS_ 1e-5f

typedef __attribute__((ext_vector_type(16))) __bf16 v16bf;
typedef __attribute__((ext_vector_type(8)))  __bf16 v8bf;
typedef __attribute__((ext_vector_type(8)))  float  v8f;

// ---------------------------------------------------------------------------
// Kernel 0a: per-code squared norms ||e||^2 -> enorm[M]
// ---------------------------------------------------------------------------
__global__ void enorm_kernel(const float* __restrict__ emb, float* __restrict__ enorm)
{
    int c = blockIdx.x * blockDim.x + threadIdx.x;
    if (c < M_) {
        const float* r = emb + (size_t)c * D_;
        float s = 0.f;
        #pragma unroll 4
        for (int d = 0; d < D_; ++d) s += r[d] * r[d];
        enorm[c] = s;
    }
}

// ---------------------------------------------------------------------------
// Kernel 0b: one-time f32 -> bf16 conversion of the embedding (L2-resident)
// ---------------------------------------------------------------------------
__global__ void ecvt_kernel(const float* __restrict__ emb, __bf16* __restrict__ ebf)
{
    int i = blockIdx.x * blockDim.x + threadIdx.x;
    if (i < M_ * D_) ebf[i] = (__bf16)emb[i];
}

// ---------------------------------------------------------------------------
// Kernel 1: fused distance GEMM (WMMA bf16) + argmin + softmax + pH accum.
// Grid: (N/16, 3 modalities), block = 256 threads (8 waves).
// Each block owns 16 rows x all 2048 codes; each wave owns a 256-col slice.
// ---------------------------------------------------------------------------
__global__ __launch_bounds__(256) void fused_dist_kernel(
    const float* __restrict__ xa, const float* __restrict__ xv,
    const float* __restrict__ xt, const __bf16* __restrict__ ebf,
    const float* __restrict__ enorm,
    int* __restrict__ idxArr, float* __restrict__ pH)
{
    __shared__ float sh_x[16][D_];     // 16 KB staged x-tile
    __shared__ float sh_xn[16];
    __shared__ float sh_wmin[8][16];
    __shared__ int   sh_warg[8][16];
    __shared__ float sh_wsum[8][16];
    __shared__ float sh_smax[16];
    __shared__ float sh_rsum[16];

    const int mod     = blockIdx.y;
    const float* x    = (mod == 0) ? xa : ((mod == 1) ? xv : xt);
    const int rowBase = blockIdx.x * 16;
    const int tid     = threadIdx.x;
    const int wave    = tid >> 5;
    const int lane    = tid & 31;
    const int n16     = lane & 15;     // A: M index / B,C: N index
    const int khalf   = lane >> 4;     // K-half selector per ISA layout

    // Stage the 16x256 x-tile in LDS
    for (int i = tid; i < 16 * D_; i += 256)
        sh_x[i >> 8][i & (D_ - 1)] = x[(size_t)rowBase * D_ + i];
    __syncthreads();
    if (tid < 16) {
        float s = 0.f;
        for (int d = 0; d < D_; ++d) { float v = sh_x[tid][d]; s += v * v; }
        sh_xn[tid] = s;
    }
    __syncthreads();

    // Build 8 A-fragments (16x32 bf16) once; held in VGPRs for all col tiles.
    // 16-bit A 16x32 layout: lane M=l&15; K = chunk*32 + (j<8?0:16) + khalf*8 + (j&7)
    v16bf afrag[8];
    #pragma unroll
    for (int ch = 0; ch < 8; ++ch) {
        #pragma unroll
        for (int j = 0; j < 16; ++j) {
            int k = ch * 32 + ((j < 8) ? 0 : 16) + khalf * 8 + (j & 7);
            afrag[ch][j] = (__bf16)sh_x[n16][k];
        }
    }

    const int colBase0 = wave * 256;
    v8f   cacc[16];                    // 16 x (16x16 f32 C tiles) = 128 VGPRs
    float en[16];

    #pragma unroll
    for (int ct = 0; ct < 16; ++ct) {
        const int colBase = colBase0 + ct * 16;
        const __bf16* erow = ebf + (size_t)(colBase + n16) * D_;  // B: lane = N
        v8f c = {0.f, 0.f, 0.f, 0.f, 0.f, 0.f, 0.f, 0.f};
        #pragma unroll
        for (int ch = 0; ch < 8; ++ch) {
            // Two contiguous 8xbf16 (16B) runs per the K-half fragment layout
            const int k0 = ch * 32 + khalf * 8;
            const int k1 = k0 + 16;
            v8bf lo = *(const v8bf*)(erow + k0);
            v8bf hi = *(const v8bf*)(erow + k1);
            v16bf b;
            #pragma unroll
            for (int j = 0; j < 8; ++j) { b[j] = lo[j]; b[j + 8] = hi[j]; }
            c = __builtin_amdgcn_wmma_f32_16x16x32_bf16(
                    false, afrag[ch], false, b, (short)0, c, false, false);
        }
        cacc[ct] = c;
        en[ct]   = enorm[colBase + n16];
    }

    // C readout: lane holds rows m = khalf*8 + r (r = VGPR index), col = colBase+ct*16+n16.
    float xn[8];
    #pragma unroll
    for (int r = 0; r < 8; ++r) xn[r] = sh_xn[khalf * 8 + r];

    // ---- per-row min + argmin over this wave's 256 columns ----
    float rmin[8]; int rarg[8];
    #pragma unroll
    for (int r = 0; r < 8; ++r) { rmin[r] = 3.4e38f; rarg[r] = 0; }
    #pragma unroll
    for (int ct = 0; ct < 16; ++ct) {
        int col = colBase0 + ct * 16 + n16;
        #pragma unroll
        for (int r = 0; r < 8; ++r) {
            float d = xn[r] + en[ct] - 2.0f * cacc[ct][r];
            if (d < rmin[r]) { rmin[r] = d; rarg[r] = col; }
        }
    }
    #pragma unroll
    for (int off = 1; off < 16; off <<= 1) {
        #pragma unroll
        for (int r = 0; r < 8; ++r) {
            float om = __shfl_xor(rmin[r], off, 32);
            int   oa = __shfl_xor(rarg[r], off, 32);
            if (om < rmin[r] || (om == rmin[r] && oa < rarg[r])) { rmin[r] = om; rarg[r] = oa; }
        }
    }
    if (n16 == 0) {
        #pragma unroll
        for (int r = 0; r < 8; ++r) {
            sh_wmin[wave][khalf * 8 + r] = rmin[r];
            sh_warg[wave][khalf * 8 + r] = rarg[r];
        }
    }
    __syncthreads();
    if (tid < 16) {
        float bm = sh_wmin[0][tid]; int ba = sh_warg[0][tid];
        for (int w = 1; w < 8; ++w) {
            float m = sh_wmin[w][tid]; int a = sh_warg[w][tid];
            if (m < bm || (m == bm && a < ba)) { bm = m; ba = a; }
        }
        idxArr[mod * N_ + rowBase + tid] = ba;
        sh_smax[tid] = -sqrtf(fmaxf(bm, 1e-12f));   // softmax row max of -sqrt(dist)
    }
    __syncthreads();

    // ---- per-row exp-sum ----
    float smax[8];
    #pragma unroll
    for (int r = 0; r < 8; ++r) smax[r] = sh_smax[khalf * 8 + r];
    float rsum[8];
    #pragma unroll
    for (int r = 0; r < 8; ++r) rsum[r] = 0.f;
    #pragma unroll
    for (int ct = 0; ct < 16; ++ct) {
        #pragma unroll
        for (int r = 0; r < 8; ++r) {
            float d = xn[r] + en[ct] - 2.0f * cacc[ct][r];
            float s = -sqrtf(fmaxf(d, 1e-12f));
            rsum[r] += __expf(s - smax[r]);
        }
    }
    #pragma unroll
    for (int off = 1; off < 16; off <<= 1)
        #pragma unroll
        for (int r = 0; r < 8; ++r) rsum[r] += __shfl_xor(rsum[r], off, 32);
    if (n16 == 0)
        for (int r = 0; r < 8; ++r) sh_wsum[wave][khalf * 8 + r] = rsum[r];
    __syncthreads();
    if (tid < 16) {
        float s = 0.f;
        for (int w = 0; w < 8; ++w) s += sh_wsum[w][tid];
        sh_rsum[tid] = s;
    }
    __syncthreads();
    float rinv[8];
    #pragma unroll
    for (int r = 0; r < 8; ++r) rinv[r] = 1.0f / sh_rsum[khalf * 8 + r];

    // ---- pH accumulation: all 16 rows belong to the same batch b ----
    const int b = rowBase / T_;
    float* ph = pH + ((size_t)mod * B_ + b) * M_;
    const float invT = 1.0f / (float)T_;
    #pragma unroll
    for (int ct = 0; ct < 16; ++ct) {
        float colsum = 0.f;
        #pragma unroll
        for (int r = 0; r < 8; ++r) {
            float d = xn[r] + en[ct] - 2.0f * cacc[ct][r];
            float s = -sqrtf(fmaxf(d, 1e-12f));
            colsum += __expf(s - smax[r]) * rinv[r];
        }
        colsum += __shfl_xor(colsum, 16, 32);       // fold the two row halves (same col)
        if (khalf == 0)
            atomicAdd(&ph[colBase0 + ct * 16 + n16], colsum * invT);
    }
}

// ---------------------------------------------------------------------------
// Kernel 2: straight-through quantized outputs (pure gather, float4)
// ---------------------------------------------------------------------------
__global__ void quant_kernel(const float* __restrict__ emb,
                             const int* __restrict__ idxArr,
                             float* __restrict__ out)
{
    size_t e = (size_t)blockIdx.x * blockDim.x + threadIdx.x;
    const size_t total = (size_t)3 * N_ * (D_ / 4);
    if (e >= total) return;
    int    d4 = (int)(e % (D_ / 4));
    size_t n3 = e / (D_ / 4);              // 0 .. 3*N
    int    mod = (int)(n3 / N_);
    int    n   = (int)(n3 % N_);
    int    code = idxArr[mod * N_ + n];
    const float4* src = (const float4*)(emb + (size_t)code * D_);
    ((float4*)out)[e] = src[d4];
}

// ---------------------------------------------------------------------------
// Kernel 3: per-(modality,batch) mode of code indices over T (ties -> smallest)
// ---------------------------------------------------------------------------
__global__ __launch_bounds__(256) void mode_kernel(const int* __restrict__ idxArr,
                                                   int* __restrict__ modes)
{
    __shared__ int hist[M_];
    __shared__ int sc[256];
    __shared__ int si[256];
    const int b = blockIdx.x, mod = blockIdx.y, t = threadIdx.x;
    for (int i = t; i < M_; i += 256) hist[i] = 0;
    __syncthreads();
    atomicAdd(&hist[idxArr[mod * N_ + b * T_ + t]], 1);
    __syncthreads();
    int bc = -1, bi = 0;
    for (int i = t * 8; i < t * 8 + 8; ++i) {
        int c = hist[i];
        if (c > bc) { bc = c; bi = i; }
    }
    sc[t] = bc; si[t] = bi;
    __syncthreads();
    for (int s = 128; s > 0; s >>= 1) {
        if (t < s) {
            if (sc[t + s] > sc[t] || (sc[t + s] == sc[t] && si[t + s] < si[t])) {
                sc[t] = sc[t + s]; si[t] = si[t + s];
            }
        }
        __syncthreads();
    }
    if (t == 0) modes[mod * B_ + b] = si[0];
}

// ---------------------------------------------------------------------------
// Kernel 4: lp = log(pH + 1e-10)
// ---------------------------------------------------------------------------
__global__ void logph_kernel(const float* __restrict__ pH, float* __restrict__ lp)
{
    int i = blockIdx.x * blockDim.x + threadIdx.x;
    if (i < 3 * B_ * M_) lp[i] = __logf(pH[i] + 1e-10f);
}

// ---------------------------------------------------------------------------
// Kernel 5: cross-modal code matching loss per pair (3 blocks)
// ---------------------------------------------------------------------------
__global__ __launch_bounds__(256) void lcmcm_kernel(const float* __restrict__ pH,
                                                    const float* __restrict__ lp,
                                                    float* __restrict__ out_losses)
{
    __shared__ float S[64][64];        // 16 KB
    __shared__ float red[256];
    const int pair = blockIdx.x;
    const int pm[3] = {0, 0, 2};
    const int qm[3] = {1, 2, 1};
    const float* p   = pH + (size_t)pm[pair] * B_ * M_;
    const float* q   = pH + (size_t)qm[pair] * B_ * M_;
    const float* lq_ = lp + (size_t)qm[pair] * B_ * M_;
    const float* lp_ = lp + (size_t)pm[pair] * B_ * M_;
    const int t = threadIdx.x;

    for (int e = t; e < 64 * 64; e += 256) {
        int i = e >> 6, j = e & 63;
        const float* pi  = p   + (size_t)i * M_;
        const float* qi  = q   + (size_t)i * M_;
        const float* lqj = lq_ + (size_t)j * M_;
        const float* lpj = lp_ + (size_t)j * M_;
        float s = 0.f;
        #pragma unroll 4
        for (int m = 0; m < M_; ++m) s += pi[m] * lqj[m] + qi[m] * lpj[m];
        S[i][j] = s;
    }
    __syncthreads();
    // Smax = max(-S) = -min(S)
    float mn = 3.4e38f;
    for (int e = t; e < 4096; e += 256) mn = fminf(mn, S[e >> 6][e & 63]);
    red[t] = mn; __syncthreads();
    for (int s = 128; s > 0; s >>= 1) {
        if (t < s) red[t] = fminf(red[t], red[t + s]);
        __syncthreads();
    }
    const float Smax = -red[0];
    __syncthreads();
    if (t < 64) {
        float rs = 0.f;
        for (int j = 0; j < 64; ++j) rs += __expf(S[t][j] + Smax);
        float diag = __expf(S[t][t] + Smax);
        red[t] = __logf(diag / (rs + EPS_));
    }
    __syncthreads();
    if (t == 0) {
        float s = 0.f;
        for (int i = 0; i < 64; ++i) s += red[i];
        out_losses[pair] = -s / 64.0f;
    }
}

// ---------------------------------------------------------------------------
// Kernel 6: equal_num = sum_b (a_mode==v_mode && a_mode==t_mode)
// ---------------------------------------------------------------------------
__global__ void equal_kernel(const int* __restrict__ modes, float* __restrict__ outv)
{
    __shared__ int cnt[64];
    const int t = threadIdx.x;
    int a = modes[t], v = modes[B_ + t], tt = modes[2 * B_ + t];
    cnt[t] = (a == v && a == tt) ? 1 : 0;
    __syncthreads();
    for (int s = 32; s > 0; s >>= 1) {
        if (t < s) cnt[t] += cnt[t + s];
        __syncthreads();
    }
    if (t == 0) outv[0] = (float)cnt[0];
}

// ---------------------------------------------------------------------------
static inline size_t alignUp(size_t x) { return (x + 255) & ~(size_t)255; }

extern "C" void kernel_launch(void* const* d_in, const int* in_sizes, int n_in,
                              void* d_out, int out_size, void* d_ws, size_t ws_size,
                              hipStream_t stream)
{
    (void)in_sizes; (void)n_in; (void)out_size; (void)ws_size;
    const float* xa  = (const float*)d_in[0];
    const float* xv  = (const float*)d_in[1];
    const float* xt  = (const float*)d_in[2];
    const float* emb = (const float*)d_in[3];
    float* out = (float*)d_out;

    char* ws = (char*)d_ws;
    float*  enorm  = (float*)ws;   ws += alignUp((size_t)M_ * 4);
    __bf16* ebf    = (__bf16*)ws;  ws += alignUp((size_t)M_ * D_ * 2);
    int*    idxArr = (int*)ws;     ws += alignUp((size_t)3 * N_ * 4);
    float*  pH     = (float*)ws;   ws += alignUp((size_t)3 * B_ * M_ * 4);
    float*  lp     = (float*)ws;   ws += alignUp((size_t)3 * B_ * M_ * 4);
    int*    modes  = (int*)ws;     ws += alignUp((size_t)3 * B_ * 4);

    hipMemsetAsync(pH, 0, (size_t)3 * B_ * M_ * 4, stream);

    enorm_kernel<<<(M_ + 255) / 256, 256, 0, stream>>>(emb, enorm);
    ecvt_kernel<<<(M_ * D_ + 255) / 256, 256, 0, stream>>>(emb, ebf);

    fused_dist_kernel<<<dim3(N_ / 16, 3), 256, 0, stream>>>(
        xa, xv, xt, ebf, enorm, idxArr, pH);

    quant_kernel<<<(int)(((size_t)3 * N_ * (D_ / 4) + 255) / 256), 256, 0, stream>>>(
        emb, idxArr, out);

    mode_kernel<<<dim3(B_, 3), 256, 0, stream>>>(idxArr, modes);

    logph_kernel<<<(3 * B_ * M_ + 255) / 256, 256, 0, stream>>>(pH, lp);

    lcmcm_kernel<<<3, 256, 0, stream>>>(pH, lp, out + (size_t)3 * N_ * D_);

    equal_kernel<<<1, 64, 0, stream>>>(modes, out + (size_t)3 * N_ * D_ + 3);
}
